// FullPageHTRDecoder_6425271075317
// MI455X (gfx1250) — matrix operand order repaired and checked by reading, not compiled, not run on hardware
//
#include <hip/hip_runtime.h>

typedef __attribute__((ext_vector_type(16))) _Float16 v16h;
typedef __attribute__((ext_vector_type(8)))  float    v8f;

#define DEVFN static __device__ __forceinline__

// ---------------- constants ----------------
#define NLAYERS 6
#define DM      512
#define NH      8
#define HD      64
#define DFF     2048
#define BATCH   8
#define MEMLEN  256
#define T1      33
#define STEPS   32
#define VOCAB   80
#define SQRT_D  22.627416997969522f   // sqrt(512)
#define INV_SQRT_HD 0.125f            // 1/sqrt(64)

// ---------------- wave helpers ----------------
DEVFN float wave_max32(float v) {
#pragma unroll
  for (int o = 16; o; o >>= 1) v = fmaxf(v, __shfl_xor(v, o, 32));
  return v;
}
DEVFN float wave_sum32(float v) {
#pragma unroll
  for (int o = 16; o; o >>= 1) v += __shfl_xor(v, o, 32);
  return v;
}

// A-fragment element address for 16x32 f16 A tiles (per WMMA VGPR layout table):
// lanes 0-15: K=0..7 (V0-3), K=16..23 (V4-7); lanes 16-31: K=8..15, K=24..31.
DEVFN int packA_idx(int m, int k) {
  int tile = k >> 5, kk = k & 31;
  int lane = m + 16 * ((kk >> 3) & 1);
  int hIdx = ((kk & 7) >> 1) * 2 + (kk & 1) + 8 * (kk >> 4);
  return tile * 512 + lane * 16 + hIdx;
}

// ---------------- packing kernels ----------------
// Pack fp32 row-major [mtiles*16, kTiles*32] into WMMA A-fragment f16 tiles.
// grid.x = mtiles*kTiles, block = 32
__global__ __launch_bounds__(32)
void pack_a(const float* __restrict__ src, _Float16* __restrict__ dst, int kTiles) {
  int lane = threadIdx.x;
  int blk  = blockIdx.x;
  int mt = blk / kTiles, kt = blk % kTiles;
  int mloc = lane & 15, laneHi = lane >> 4;
  int K = kTiles * 32;
#pragma unroll
  for (int hi = 0; hi < 16; ++hi) {
    int vg = hi >> 1, half = hi & 1;
    int kk = 2 * (vg & 3) + half + 16 * (vg >> 2) + 8 * laneHi;
    int m = mt * 16 + mloc;
    int k = kt * 32 + kk;
    dst[(size_t)blk * 512 + lane * 16 + hi] = (_Float16)src[(size_t)m * K + k];
  }
}

// Pack weight W[N,K] (row-major) as B-fragments: B[k,n] = W[n,k].
// B 32x16 tile: lanes 0-15 hold K=0..7,16..23 (col=lane), lanes 16-31 hold K=8..15,24..31.
// grid.x = nTiles*kTiles, block = 32
__global__ __launch_bounds__(32)
void pack_b(const float* __restrict__ W, _Float16* __restrict__ dst, int kTiles, int K) {
  int lane = threadIdx.x;
  int blk  = blockIdx.x;
  int ntile = blk / kTiles, kt = blk % kTiles;
  int n = ntile * 16 + (lane & 15);
  int laneHi = lane >> 4;
#pragma unroll
  for (int hi = 0; hi < 16; ++hi) {
    int kk = (hi & 7) + 8 * laneHi + 16 * (hi >> 3);
    int k  = kt * 32 + kk;
    dst[(size_t)blk * 512 + lane * 16 + hi] = (_Float16)W[(size_t)n * K + k];
  }
}

// ---------------- GEMM (one wave = one 16x16 C tile, K-loop of WMMA) ----------------
enum { GM_PLAIN = 0, GM_QKV = 1, GM_GELU_PACK = 2 };

template <int MODE>
__global__ __launch_bounds__(32)
void gemm16(const _Float16* __restrict__ Ap,   // packed A, kTiles tiles
            const _Float16* __restrict__ Bp,   // packed B, offset to this layer's ntile base
            const float*    __restrict__ bias, // offset to col base
            int kTiles,
            float*    __restrict__ outF,       // PLAIN: [16,ldN]; QKV: q buffer [16,512]
            int ldN,
            _Float16* __restrict__ outPacked,  // GELU_PACK: A-fragment dest
            float* __restrict__ kcache, float* __restrict__ vcache, // QKV
            int layer, int t) {
  const int lane = threadIdx.x;
  const int nt   = blockIdx.x;
  v8f acc = {0.f, 0.f, 0.f, 0.f, 0.f, 0.f, 0.f, 0.f};
  const _Float16* aPtr = Ap + (size_t)lane * 16;
  const _Float16* bPtr = Bp + ((size_t)nt * kTiles) * 512 + (size_t)lane * 16;
  for (int kt = 0; kt < kTiles; ++kt) {
    v16h a = *(const v16h*)(aPtr + (size_t)kt * 512);
    v16h b = *(const v16h*)(bPtr + (size_t)kt * 512);
    acc = __builtin_amdgcn_wmma_f32_16x16x32_f16(false, a, false, b, (short)0, acc,
                                                 false, false);
  }
  const int n = lane & 15;
  const int mBase = (lane >> 4) * 8;
  const int col = nt * 16 + n;
  const float bv = bias[col];
#pragma unroll
  for (int v = 0; v < 8; ++v) {
    const int m = mBase + v;
    float c = acc[v] + bv;
    if constexpr (MODE == GM_PLAIN) {
      outF[(size_t)m * ldN + col] = c;
    } else if constexpr (MODE == GM_QKV) {
      if (col < DM) {
        outF[(size_t)m * DM + col] = c;                       // q
      } else if (m < BATCH) {
        size_t row = (((size_t)(layer * BATCH + m)) * T1 + t) * DM;
        if (col < 2 * DM) kcache[row + (col - DM)] = c;
        else              vcache[row + (col - 2 * DM)] = c;
      }
    } else { // GM_GELU_PACK : exact gelu then scatter into A-fragment layout
      float g = 0.5f * c * (1.f + erff(c * 0.70710678118654752f));
      outPacked[packA_idx(m, col)] = (_Float16)g;
    }
  }
}

// ---------------- cross-attn K/V precompute GEMM ----------------
// grid = (128 mtiles, 64 ntiles, 6 layers); A = packed memory [2048,512]
__global__ __launch_bounds__(32)
void gemm_prep_kv(const _Float16* __restrict__ memA, const _Float16* __restrict__ caInP,
                  const float* __restrict__ caInB,
                  _Float16* __restrict__ Kmem, _Float16* __restrict__ Vmem) {
  const int lane = threadIdx.x;
  const int mt = blockIdx.x, ntL = blockIdx.y, layer = blockIdx.z;
  const int kTiles = 16;
  v8f acc = {0.f, 0.f, 0.f, 0.f, 0.f, 0.f, 0.f, 0.f};
  const _Float16* aPtr = memA + ((size_t)mt * kTiles) * 512 + (size_t)lane * 16;
  const int ntGlobal = layer * 96 + 32 + ntL; // skip q-block (32 tiles) of this layer
  const _Float16* bPtr = caInP + ((size_t)ntGlobal * kTiles) * 512 + (size_t)lane * 16;
  for (int kt = 0; kt < kTiles; ++kt) {
    v16h a = *(const v16h*)(aPtr + (size_t)kt * 512);
    v16h b = *(const v16h*)(bPtr + (size_t)kt * 512);
    acc = __builtin_amdgcn_wmma_f32_16x16x32_f16(false, a, false, b, (short)0, acc,
                                                 false, false);
  }
  const int n = lane & 15;
  const int mBase = (lane >> 4) * 8;
  const int colL = DM + ntL * 16 + n; // [512,1536) within this layer's in-proj
  const float bv = caInB[layer * 3 * DM + colL];
#pragma unroll
  for (int v = 0; v < 8; ++v) {
    int mg = mt * 16 + mBase + v;
    int b = mg >> 8, j = mg & 255;
    float c = acc[v] + bv;
    size_t base = (((size_t)(layer * BATCH + b)) * MEMLEN + j) * DM;
    if (colL < 2 * DM) Kmem[base + (colL - DM)]     = (_Float16)c;
    else               Vmem[base + (colL - 2 * DM)] = (_Float16)c;
  }
}

// ---------------- attention (VALU) ----------------
__global__ __launch_bounds__(32)
void self_attn(const float* __restrict__ qsa, const float* __restrict__ kcache,
               const float* __restrict__ vcache, _Float16* __restrict__ attnPacked,
               int layer, int t) {
  const int lane = threadIdx.x;
  const int b = blockIdx.x >> 3, h = blockIdx.x & 7;
  __shared__ float qs[HD];
  __shared__ float ps[64];
  qs[lane]      = qsa[b * DM + h * HD + lane];
  qs[lane + 32] = qsa[b * DM + h * HD + lane + 32];
  __syncthreads();
  const int nk = t + 1;
  const float* Kb = kcache + (((size_t)(layer * BATCH + b)) * T1) * DM + h * HD;
  const float* Vb = vcache + (((size_t)(layer * BATCH + b)) * T1) * DM + h * HD;
  float s0 = -1e30f, s1 = -1e30f;
  if (lane < nk) {
    const float* kr = Kb + (size_t)lane * DM;
    float d = 0.f;
    for (int i = 0; i < HD; ++i) d += qs[i] * kr[i];
    s0 = d * INV_SQRT_HD;
  }
  if (lane + 32 < nk) {
    const float* kr = Kb + (size_t)(lane + 32) * DM;
    float d = 0.f;
    for (int i = 0; i < HD; ++i) d += qs[i] * kr[i];
    s1 = d * INV_SQRT_HD;
  }
  float mx = wave_max32(fmaxf(s0, s1));
  float e0 = (lane < nk)      ? __expf(s0 - mx) : 0.f;
  float e1 = (lane + 32 < nk) ? __expf(s1 - mx) : 0.f;
  float inv = 1.f / wave_sum32(e0 + e1);
  ps[lane] = e0 * inv;
  ps[lane + 32] = e1 * inv;
  __syncthreads();
  for (int dd = lane; dd < HD; dd += 32) {
    float o = 0.f;
    for (int j = 0; j < nk; ++j) o += ps[j] * Vb[(size_t)j * DM + dd];
    attnPacked[packA_idx(b, h * HD + dd)] = (_Float16)o;
  }
}

__global__ __launch_bounds__(32)
void cross_attn(const float* __restrict__ qca, const _Float16* __restrict__ Kmem,
                const _Float16* __restrict__ Vmem, _Float16* __restrict__ attnPacked,
                int layer) {
  const int lane = threadIdx.x;
  const int b = blockIdx.x >> 3, h = blockIdx.x & 7;
  __shared__ float qs[HD];
  __shared__ float ps[MEMLEN];
  qs[lane]      = qca[b * DM + h * HD + lane];
  qs[lane + 32] = qca[b * DM + h * HD + lane + 32];
  __syncthreads();
  const _Float16* Kb = Kmem + (((size_t)(layer * BATCH + b)) * MEMLEN) * DM + h * HD;
  const _Float16* Vb = Vmem + (((size_t)(layer * BATCH + b)) * MEMLEN) * DM + h * HD;
  float sc[8];
  float mx = -1e30f;
#pragma unroll
  for (int r = 0; r < 8; ++r) {
    int j = lane + 32 * r;
    const _Float16* kr = Kb + (size_t)j * DM;
    float d = 0.f;
    for (int i = 0; i < HD; ++i) d += qs[i] * (float)kr[i];
    sc[r] = d * INV_SQRT_HD;
    mx = fmaxf(mx, sc[r]);
  }
  mx = wave_max32(mx);
  float sum = 0.f;
#pragma unroll
  for (int r = 0; r < 8; ++r) { sc[r] = __expf(sc[r] - mx); sum += sc[r]; }
  float inv = 1.f / wave_sum32(sum);
#pragma unroll
  for (int r = 0; r < 8; ++r) ps[lane + 32 * r] = sc[r] * inv;
  __syncthreads();
  for (int dd = lane; dd < HD; dd += 32) {
    float o = 0.f;
    for (int j = 0; j < MEMLEN; ++j) o += ps[j] * (float)Vb[(size_t)j * DM + dd];
    attnPacked[packA_idx(b, h * HD + dd)] = (_Float16)o;
  }
}

// ---------------- residual + LayerNorm (+ repack A fragment) ----------------
__global__ __launch_bounds__(256)
void add_ln(float* __restrict__ x, const float* __restrict__ tmp,
            const float* __restrict__ g, const float* __restrict__ be,
            _Float16* __restrict__ xPacked) {
  const int b = blockIdx.x;     // row 0..7
  const int tid = threadIdx.x;
  __shared__ float red[256];
  __shared__ float s_mu, s_rstd;
  float v0 = x[b * DM + tid]       + tmp[b * DM + tid];
  float v1 = x[b * DM + tid + 256] + tmp[b * DM + tid + 256];
  red[tid] = v0 + v1;
  __syncthreads();
  for (int o = 128; o; o >>= 1) { if (tid < o) red[tid] += red[tid + o]; __syncthreads(); }
  if (tid == 0) s_mu = red[0] * (1.f / DM);
  __syncthreads();
  float mu = s_mu;
  red[tid] = (v0 - mu) * (v0 - mu) + (v1 - mu) * (v1 - mu);
  __syncthreads();
  for (int o = 128; o; o >>= 1) { if (tid < o) red[tid] += red[tid + o]; __syncthreads(); }
  if (tid == 0) s_rstd = rsqrtf(red[0] * (1.f / DM) + 1e-5f);
  __syncthreads();
  float rs = s_rstd;
  {
    int k = tid;
    float y = (v0 - mu) * rs * g[k] + be[k];
    x[b * DM + k] = y;
    xPacked[packA_idx(b, k)] = (_Float16)y;
  }
  {
    int k = tid + 256;
    float y = (v1 - mu) * rs * g[k] + be[k];
    x[b * DM + k] = y;
    xPacked[packA_idx(b, k)] = (_Float16)y;
  }
}

// ---------------- init / argmax-advance / finalize ----------------
__global__ __launch_bounds__(512)
void init_x(const float* __restrict__ pe, const float* __restrict__ emb,
            float* __restrict__ x, int* __restrict__ sampled) {
  int row = blockIdx.x, k = threadIdx.x;
  float v = (row < BATCH) ? (pe[k] + emb[2 * DM + k] * SQRT_D) : 0.f; // SOS=2
  x[row * DM + k] = v;
  if (row == 0 && k < BATCH) sampled[k * T1 + 0] = 2;
}

__global__ __launch_bounds__(256)
void step_advance(const float* __restrict__ logits, float* __restrict__ dOut,
                  const float* __restrict__ pe, const float* __restrict__ emb,
                  float* __restrict__ x, int* __restrict__ sampled, int t) {
  __shared__ int preds[BATCH];
  const int tid = threadIdx.x;
  if (tid < BATCH) {
    const float* lr = logits + tid * VOCAB;
    int best = 0; float bv = lr[0];
    for (int v = 1; v < VOCAB; ++v) { float f = lr[v]; if (f > bv) { bv = f; best = v; } }
    preds[tid] = best;
    sampled[tid * T1 + t + 1] = best;
  }
  for (int i = tid; i < BATCH * VOCAB; i += 256) {
    int b = i / VOCAB, v = i % VOCAB;
    dOut[(size_t)b * STEPS * VOCAB + (size_t)t * VOCAB + v] = logits[b * VOCAB + v];
  }
  __syncthreads();
  for (int i = tid; i < BATCH * DM; i += 256) {
    int b = i >> 9, k = i & (DM - 1);
    x[b * DM + k] = pe[(t + 2) * DM + k] + emb[preds[b] * DM + k] * SQRT_D;
  }
}

__global__ __launch_bounds__(32)
void finalize_sampled(const int* __restrict__ sampled, float* __restrict__ dOut) {
  int b = threadIdx.x;
  if (b < BATCH) {
    int eos = 0;
    for (int i = 0; i < T1; ++i) { if (sampled[b * T1 + i] == 1) { eos = i; break; } }
    for (int i = 0; i < T1; ++i) {
      int tok = sampled[b * T1 + i];
      dOut[BATCH * STEPS * VOCAB + b * T1 + i] = (i <= eos) ? 0.f : (float)tok;
    }
  }
}

// ---------------- host orchestration ----------------
extern "C" void kernel_launch(void* const* d_in, const int* in_sizes, int n_in,
                              void* d_out, int out_size, void* d_ws, size_t ws_size,
                              hipStream_t stream) {
  (void)in_sizes; (void)n_in; (void)out_size; (void)ws_size;
  const float* memory   = (const float*)d_in[0];
  const float* emb      = (const float*)d_in[1];
  const float* pe       = (const float*)d_in[2];
  const float* sa_in_w  = (const float*)d_in[3];
  const float* sa_in_b  = (const float*)d_in[4];
  const float* sa_out_w = (const float*)d_in[5];
  const float* sa_out_b = (const float*)d_in[6];
  const float* ca_in_w  = (const float*)d_in[7];
  const float* ca_in_b  = (const float*)d_in[8];
  const float* ca_out_w = (const float*)d_in[9];
  const float* ca_out_b = (const float*)d_in[10];
  const float* ln_w     = (const float*)d_in[11];
  const float* ln_b     = (const float*)d_in[12];
  const float* ff1_w    = (const float*)d_in[13];
  const float* ff1_b    = (const float*)d_in[14];
  const float* ff2_w    = (const float*)d_in[15];
  const float* ff2_b    = (const float*)d_in[16];
  const float* clf_w    = (const float*)d_in[17];
  const float* clf_b    = (const float*)d_in[18];
  float* out = (float*)d_out;

  // ---- workspace carve-up (256B aligned) ----
  size_t off = 0;
  char* wsb = (char*)d_ws;
  auto carve = [&](size_t bytes) -> char* {
    char* p = wsb + off;
    off = (off + bytes + 255) & ~(size_t)255;
    return p;
  };
  _Float16* saInP  = (_Float16*)carve((size_t)NLAYERS * 1536 * DM * 2);
  _Float16* saOutP = (_Float16*)carve((size_t)NLAYERS * DM * DM * 2);
  _Float16* caInP  = (_Float16*)carve((size_t)NLAYERS * 1536 * DM * 2);
  _Float16* caOutP = (_Float16*)carve((size_t)NLAYERS * DM * DM * 2);
  _Float16* ff1P   = (_Float16*)carve((size_t)NLAYERS * DFF * DM * 2);
  _Float16* ff2P   = (_Float16*)carve((size_t)NLAYERS * DM * DFF * 2);
  _Float16* clfP   = (_Float16*)carve((size_t)VOCAB * DM * 2);
  _Float16* memA   = (_Float16*)carve((size_t)BATCH * MEMLEN * DM * 2);
  _Float16* KmemH  = (_Float16*)carve((size_t)NLAYERS * BATCH * MEMLEN * DM * 2);
  _Float16* VmemH  = (_Float16*)carve((size_t)NLAYERS * BATCH * MEMLEN * DM * 2);
  float*    kcache = (float*)carve((size_t)NLAYERS * BATCH * T1 * DM * 4);
  float*    vcache = (float*)carve((size_t)NLAYERS * BATCH * T1 * DM * 4);
  float*    x      = (float*)carve((size_t)16 * DM * 4);
  _Float16* xPacked= (_Float16*)carve((size_t)16 * DM * 2);
  float*    qsa    = (float*)carve((size_t)16 * DM * 4);
  _Float16* attnP  = (_Float16*)carve((size_t)16 * DM * 2);
  float*    qca    = (float*)carve((size_t)16 * DM * 4);
  _Float16* attn2P = (_Float16*)carve((size_t)16 * DM * 2);
  _Float16* hP     = (_Float16*)carve((size_t)16 * DFF * 2);
  float*    tmp    = (float*)carve((size_t)16 * DM * 4);
  float*    logits = (float*)carve((size_t)16 * VOCAB * 4);
  int*      sampled= (int*)carve((size_t)BATCH * T1 * 4);

  // ---- one-time prep: pack weights, pack memory, precompute cross K/V ----
  pack_b<<<6 * 96 * 16, 32, 0, stream>>>(sa_in_w,  saInP,  16, DM);
  pack_b<<<6 * 32 * 16, 32, 0, stream>>>(sa_out_w, saOutP, 16, DM);
  pack_b<<<6 * 96 * 16, 32, 0, stream>>>(ca_in_w,  caInP,  16, DM);
  pack_b<<<6 * 32 * 16, 32, 0, stream>>>(ca_out_w, caOutP, 16, DM);
  pack_b<<<6 * 128 * 16, 32, 0, stream>>>(ff1_w,   ff1P,   16, DM);
  pack_b<<<6 * 32 * 64, 32, 0, stream>>>(ff2_w,    ff2P,   64, DFF);
  pack_b<<<5 * 16, 32, 0, stream>>>(clf_w,         clfP,   16, DM);
  pack_a<<<128 * 16, 32, 0, stream>>>(memory, memA, 16);
  gemm_prep_kv<<<dim3(128, 64, 6), 32, 0, stream>>>(memA, caInP, ca_in_b, KmemH, VmemH);
  init_x<<<16, 512, 0, stream>>>(pe, emb, x, sampled);

  // per-layer packed-B strides (halves)
  const size_t strInProj = (size_t)96 * 16 * 512;   // 1536-wide, K=512
  const size_t strSq     = (size_t)32 * 16 * 512;   // 512-wide,  K=512
  const size_t strFf1    = (size_t)128 * 16 * 512;  // 2048-wide, K=512
  const size_t strFf2    = (size_t)32 * 64 * 512;   // 512-wide,  K=2048

  for (int t = 0; t < STEPS; ++t) {
    // pack current input row block (x was written by init/step_advance)
    pack_a<<<16, 32, 0, stream>>>(x, xPacked, 16);
    for (int l = 0; l < NLAYERS; ++l) {
      // --- self-attention ---
      gemm16<GM_QKV><<<96, 32, 0, stream>>>(xPacked, saInP + (size_t)l * strInProj,
          sa_in_b + l * 1536, 16, qsa, DM, nullptr, kcache, vcache, l, t);
      self_attn<<<64, 32, 0, stream>>>(qsa, kcache, vcache, attnP, l, t);
      gemm16<GM_PLAIN><<<32, 32, 0, stream>>>(attnP, saOutP + (size_t)l * strSq,
          sa_out_b + l * DM, 16, tmp, DM, nullptr, nullptr, nullptr, l, t);
      add_ln<<<8, 256, 0, stream>>>(x, tmp, ln_w + (l * 3 + 0) * DM,
                                    ln_b + (l * 3 + 0) * DM, xPacked);
      // --- cross-attention (K/V precomputed) ---
      gemm16<GM_PLAIN><<<32, 32, 0, stream>>>(xPacked, caInP + (size_t)l * strInProj,
          ca_in_b + l * 1536, 16, qca, DM, nullptr, nullptr, nullptr, l, t);
      cross_attn<<<64, 32, 0, stream>>>(qca, KmemH, VmemH, attn2P, l);
      gemm16<GM_PLAIN><<<32, 32, 0, stream>>>(attn2P, caOutP + (size_t)l * strSq,
          ca_out_b + l * DM, 16, tmp, DM, nullptr, nullptr, nullptr, l, t);
      add_ln<<<8, 256, 0, stream>>>(x, tmp, ln_w + (l * 3 + 1) * DM,
                                    ln_b + (l * 3 + 1) * DM, xPacked);
      // --- FFN ---
      gemm16<GM_GELU_PACK><<<128, 32, 0, stream>>>(xPacked, ff1P + (size_t)l * strFf1,
          ff1_b + l * DFF, 16, nullptr, 0, hP, nullptr, nullptr, l, t);
      gemm16<GM_PLAIN><<<32, 32, 0, stream>>>(hP, ff2P + (size_t)l * strFf2,
          ff2_b + l * DM, 64, tmp, DM, nullptr, nullptr, nullptr, l, t);
      add_ln<<<8, 256, 0, stream>>>(x, tmp, ln_w + (l * 3 + 2) * DM,
                                    ln_b + (l * 3 + 2) * DM, xPacked);
    }
    // --- classifier + greedy step ---
    gemm16<GM_PLAIN><<<5, 32, 0, stream>>>(xPacked, clfP, clf_b, 16, logits, VOCAB,
                                           nullptr, nullptr, nullptr, 0, t);
    step_advance<<<1, 256, 0, stream>>>(logits, out, pe, emb, x, sampled, t);
  }
  finalize_sampled<<<1, 32, 0, stream>>>(sampled, out);
}
